// RN_85169201479929
// MI455X (gfx1250) — compile-verified, matrix operand order checked by reading, not run on hardware
//
#include <hip/hip_runtime.h>
#include <hip/hip_bf16.h>

typedef __attribute__((ext_vector_type(16))) _Float16 v16h;
typedef __attribute__((ext_vector_type(8)))  _Float16 v8h;
typedef __attribute__((ext_vector_type(8)))  float    v8f;
typedef int v4i_vs __attribute__((vector_size(16)));   // matches builtin proto

#define NB 64            // batch
#define NCELL 25
#define NPAIRROWS (NB * NCELL * NCELL)   // 40000

#if __has_builtin(__builtin_amdgcn_global_load_async_to_lds_b128) && \
    __has_builtin(__builtin_amdgcn_s_wait_asynccnt)
#define USE_ASYNC_LDS 1
#else
#define USE_ASYNC_LDS 0
#endif

// ---------------------------------------------------------------------------
// CNN: direct conv (stride 2, pad 1, 3x3) + bias + relu
// ---------------------------------------------------------------------------
__global__ void conv_relu_k(const float* __restrict__ x, const float* __restrict__ w,
                            const float* __restrict__ bias, float* __restrict__ y,
                            int Cin, int Hin, int Win, int Cout, int Ho, int Wo)
{
    int idx = blockIdx.x * blockDim.x + threadIdx.x;
    int total = NB * Cout * Ho * Wo;
    if (idx >= total) return;
    int wo = idx % Wo; int t = idx / Wo;
    int ho = t % Ho;   t /= Ho;
    int co = t % Cout; int b = t / Cout;
    float acc = bias[co];
    int hi0 = ho * 2 - 1, wi0 = wo * 2 - 1;
    for (int ci = 0; ci < Cin; ++ci) {
        const float* xp = x + ((b * Cin + ci) * Hin) * Win;
        const float* wp = w + (co * Cin + ci) * 9;
#pragma unroll
        for (int kh = 0; kh < 3; ++kh) {
            int hi = hi0 + kh;
            if (hi < 0 || hi >= Hin) continue;
#pragma unroll
            for (int kw = 0; kw < 3; ++kw) {
                int wi = wi0 + kw;
                if (wi < 0 || wi >= Win) continue;
                acc += xp[hi * Win + wi] * wp[kh * 3 + kw];
            }
        }
    }
    y[idx] = fmaxf(acc, 0.0f);
}

// training-mode BN batch stats: one block per channel
__global__ void bn_stats_k(const float* __restrict__ y, float* __restrict__ mean,
                           float* __restrict__ rstd, int C, int HW)
{
    __shared__ float ss[256], sq[256];
    int c = blockIdx.x;
    int tid = threadIdx.x;
    int perC = NB * HW;
    float s = 0.f, q = 0.f;
    for (int j = tid; j < perC; j += 256) {
        int b = j / HW, p = j % HW;
        float v = y[(b * C + c) * HW + p];
        s += v; q += v * v;
    }
    ss[tid] = s; sq[tid] = q;
    __syncthreads();
    for (int st = 128; st > 0; st >>= 1) {
        if (tid < st) { ss[tid] += ss[tid + st]; sq[tid] += sq[tid + st]; }
        __syncthreads();
    }
    if (tid == 0) {
        float m = ss[0] / (float)perC;
        float var = sq[0] / (float)perC - m * m;
        mean[c] = m;
        rstd[c] = rsqrtf(var + 1e-5f);
    }
}

__global__ void bn_apply_k(float* __restrict__ y, const float* __restrict__ mean,
                           const float* __restrict__ rstd, const float* __restrict__ g,
                           const float* __restrict__ beta, int C, int HW)
{
    int idx = blockIdx.x * blockDim.x + threadIdx.x;
    int total = NB * C * HW;
    if (idx >= total) return;
    int c = (idx / HW) % C;
    y[idx] = (y[idx] - mean[c]) * rstd[c] * g[c] + beta[c];
}

// [B,24,5,5] -> feat [B,25,26] (24 channels + 2 coords, true-division coord 0)
__global__ void build_feat_k(const float* __restrict__ x5, float* __restrict__ feat)
{
    int idx = blockIdx.x * blockDim.x + threadIdx.x;
    if (idx >= NB * 25 * 26) return;
    int j = idx % 26; int t = idx / 26;
    int cell = t % 25; int b = t / 25;
    float v;
    if (j < 24)       v = x5[(b * 24 + j) * 25 + cell];
    else if (j == 24) v = ((float)cell / 5.0f - 2.0f) * 0.5f;
    else              v = ((float)(cell % 5) - 2.0f) * 0.5f;
    feat[idx] = v;
}

// ---------------------------------------------------------------------------
// g_fc1 low-rank decomposition pieces (f32, tiny)
// ---------------------------------------------------------------------------
__global__ void u_proj_k(const float* __restrict__ feat, const float* __restrict__ W1,
                         int wOff, float* __restrict__ U)
{
    int idx = blockIdx.x * blockDim.x + threadIdx.x;   // (B*25) x 256
    if (idx >= NB * 25 * 256) return;
    int o = idx % 256; int r = idx / 256;
    const float* fp = feat + r * 26;
    const float* wp = W1 + o * 702 + wOff;
    float acc = 0.f;
#pragma unroll
    for (int k = 0; k < 26; ++k) acc += fp[k] * wp[k];
    U[idx] = acc;
}

__global__ void uq_k(const float* __restrict__ feat2, const float* __restrict__ W1,
                     float* __restrict__ Uq)
{
    int idx = blockIdx.x * blockDim.x + threadIdx.x;   // B x 256
    if (idx >= NB * 256) return;
    int o = idx % 256; int b = idx / 256;
    const float* fp = feat2 + b * 650;
    const float* wp = W1 + o * 702 + 52;
    float acc = 0.f;
    for (int k = 0; k < 650; ++k) acc += fp[k] * wp[k];
    Uq[idx] = acc;
}

// h1[b,a,c,o] = relu(Uc[b,c,o] + Ua[b,a,o] + Uq[b,o] + bias[o]) -> f16
__global__ void h1_k(const float* __restrict__ Uc, const float* __restrict__ Ua,
                     const float* __restrict__ Uq, const float* __restrict__ b1,
                     _Float16* __restrict__ h)
{
    int idx = blockIdx.x * blockDim.x + threadIdx.x;   // 40000*256 = 10.24M
    if (idx >= NPAIRROWS * 256) return;
    int o = idx % 256; int n = idx / 256;
    int c = n % 25; int a = (n / 25) % 25; int b = n / 625;
    float v = Uc[(b * 25 + c) * 256 + o] + Ua[(b * 25 + a) * 256 + o]
            + Uq[b * 256 + o] + b1[o];
    h[idx] = (_Float16)fmaxf(v, 0.0f);
}

__global__ void f32_to_f16_k(const float* __restrict__ src, _Float16* __restrict__ dst, int n)
{
    int idx = blockIdx.x * blockDim.x + threadIdx.x;
    if (idx < n) dst[idx] = (_Float16)src[idx];
}

// ---------------------------------------------------------------------------
// WMMA GEMM: out[M,256] = relu(A[M,256] @ W[256,256]^T + bias)
// A row-major f16 streamed from global; W strip [64x256] f16 staged once per
// workgroup in LDS (async-to-LDS when available) and shared by 8 waves.
// grid = (ceil(M/128), 4); block = 256 threads = 8 waves; each wave owns one
// 16-row x 64-col tile: 4 f32 accumulators, K=256 -> 8 unrolled
// v_wmma_f32_16x16x32_f16 steps per accumulator.
// ---------------------------------------------------------------------------
__global__ __launch_bounds__(256)
void gemm_wmma_k(const _Float16* __restrict__ A, const _Float16* __restrict__ W,
                 const float* __restrict__ bias, _Float16* __restrict__ outH,
                 float* __restrict__ outF, int M)
{
    constexpr int LDW = 264;                 // 256 + 8-half pad: 528B rows (16B aligned)
    __shared__ __align__(16) _Float16 sW[64 * LDW];

    const int tid  = threadIdx.x;
    const int lane = tid & 31;
    const int wave = tid >> 5;
    const int cn   = blockIdx.y * 64;        // this block's 64-col strip

    // ---- cooperative stage of W strip [64 x 256] halves into LDS ----
#pragma unroll
    for (int i = 0; i < 8; ++i) {
        int c   = tid + i * 256;             // 2048 x 16B chunks
        int row = c >> 5;                    // 32 chunks per 256-half row
        int off = (c & 31) << 3;
        const _Float16* gp = W + (cn + row) * 256 + off;
        _Float16*       lp = &sW[row * LDW + off];
#if USE_ASYNC_LDS
        __builtin_amdgcn_global_load_async_to_lds_b128(
            (__attribute__((address_space(1))) v4i_vs*)gp,
            (__attribute__((address_space(3))) v4i_vs*)lp, 0, 0);
#else
        *(v8h*)lp = *(const v8h*)gp;
#endif
    }
#if USE_ASYNC_LDS
    __builtin_amdgcn_s_wait_asynccnt(0);
#endif
    __syncthreads();

    const int mTile = blockIdx.x * 8 + wave;
    if (mTile * 16 >= M) return;
    const int rm   = mTile * 16;
    const int lrow = lane & 15;
    const int koff = (lane >> 4) << 3;       // 0 or 8 (fragment K-half select)

    const _Float16* aBase = A + (rm + lrow) * 256 + koff;
    const _Float16* b0p = &sW[(0 * 16 + lrow) * LDW + koff];
    const _Float16* b1p = &sW[(1 * 16 + lrow) * LDW + koff];
    const _Float16* b2p = &sW[(2 * 16 + lrow) * LDW + koff];
    const _Float16* b3p = &sW[(3 * 16 + lrow) * LDW + koff];

    v8f acc0 = {}; v8f acc1 = {}; v8f acc2 = {}; v8f acc3 = {};

#pragma unroll
    for (int k0 = 0; k0 < 256; k0 += 32) {
        v8h alo = *(const v8h*)(aBase + k0);
        v8h ahi = *(const v8h*)(aBase + k0 + 16);
        v16h af = __builtin_shufflevector(alo, ahi,
            0,1,2,3,4,5,6,7,8,9,10,11,12,13,14,15);

        v8h b0l = *(const v8h*)(b0p + k0);
        v8h b0h = *(const v8h*)(b0p + k0 + 16);
        v16h bf0 = __builtin_shufflevector(b0l, b0h,
            0,1,2,3,4,5,6,7,8,9,10,11,12,13,14,15);
        acc0 = __builtin_amdgcn_wmma_f32_16x16x32_f16(false, af, false, bf0,
                                                      (short)0, acc0, false, false);

        v8h b1l = *(const v8h*)(b1p + k0);
        v8h b1h = *(const v8h*)(b1p + k0 + 16);
        v16h bf1 = __builtin_shufflevector(b1l, b1h,
            0,1,2,3,4,5,6,7,8,9,10,11,12,13,14,15);
        acc1 = __builtin_amdgcn_wmma_f32_16x16x32_f16(false, af, false, bf1,
                                                      (short)0, acc1, false, false);

        v8h b2l = *(const v8h*)(b2p + k0);
        v8h b2h = *(const v8h*)(b2p + k0 + 16);
        v16h bf2 = __builtin_shufflevector(b2l, b2h,
            0,1,2,3,4,5,6,7,8,9,10,11,12,13,14,15);
        acc2 = __builtin_amdgcn_wmma_f32_16x16x32_f16(false, af, false, bf2,
                                                      (short)0, acc2, false, false);

        v8h b3l = *(const v8h*)(b3p + k0);
        v8h b3h = *(const v8h*)(b3p + k0 + 16);
        v16h bf3 = __builtin_shufflevector(b3l, b3h,
            0,1,2,3,4,5,6,7,8,9,10,11,12,13,14,15);
        acc3 = __builtin_amdgcn_wmma_f32_16x16x32_f16(false, af, false, bf3,
                                                      (short)0, acc3, false, false);
    }

    // C/D layout: VGPR r, lanes 0-15 -> M=r, lanes 16-31 -> M=r+8; N = lane&15
    const int mrow = rm + ((lane >> 4) << 3);
#pragma unroll
    for (int j = 0; j < 4; ++j) {
        v8f acc = (j == 0) ? acc0 : (j == 1) ? acc1 : (j == 2) ? acc2 : acc3;
        int n = cn + j * 16 + lrow;
        float bn = bias[n];
#pragma unroll
        for (int r = 0; r < 8; ++r) {
            float v = fmaxf(acc[r] + bn, 0.0f);
            int m = mrow + r;
            if (outH) outH[m * 256 + n] = (_Float16)v;
            if (outF) outF[m * 256 + n] = v;
        }
    }
}

// sum pool over 625 pairs: xg[b,o] = sum_p h4[(b*625+p)*256 + o]
__global__ void pool_k(const float* __restrict__ h4, float* __restrict__ xg)
{
    int idx = blockIdx.x * blockDim.x + threadIdx.x;   // B x 256
    if (idx >= NB * 256) return;
    int o = idx % 256; int b = idx / 256;
    const float* p = h4 + (b * 625) * 256 + o;
    float s = 0.f;
    for (int i = 0; i < 625; ++i) s += p[i * 256];
    xg[idx] = s;
}

// small f32 FC: out[M,O] = act(in[M,K] @ W[O,K]^T + b)
__global__ void fc_k(const float* __restrict__ in, const float* __restrict__ W,
                     const float* __restrict__ bias, float* __restrict__ out,
                     int M, int K, int O, int relu)
{
    int idx = blockIdx.x * blockDim.x + threadIdx.x;
    if (idx >= M * O) return;
    int o = idx % O; int m = idx / O;
    const float* ip = in + m * K;
    const float* wp = W + o * K;
    float acc = bias[o];
    for (int k = 0; k < K; ++k) acc += ip[k] * wp[k];
    out[idx] = relu ? fmaxf(acc, 0.0f) : acc;
}

// softmax over groups of 8, then lower-clip 0.001 (upper clip at global max is identity)
__global__ void softmax8_k(const float* __restrict__ y, float* __restrict__ out)
{
    int g = blockIdx.x * blockDim.x + threadIdx.x;     // 64*16 groups
    if (g >= NB * 16) return;
    const float* p = y + g * 8;
    float m = p[0];
#pragma unroll
    for (int i = 1; i < 8; ++i) m = fmaxf(m, p[i]);
    float e[8]; float s = 0.f;
#pragma unroll
    for (int i = 0; i < 8; ++i) { e[i] = expf(p[i] - m); s += e[i]; }
    float inv = 1.0f / s;
#pragma unroll
    for (int i = 0; i < 8; ++i) out[g * 8 + i] = fmaxf(e[i] * inv, 0.001f);
}

// ---------------------------------------------------------------------------
extern "C" void kernel_launch(void* const* d_in, const int* in_sizes, int n_in,
                              void* d_out, int out_size, void* d_ws, size_t ws_size,
                              hipStream_t stream)
{
    (void)in_sizes; (void)n_in; (void)out_size; (void)ws_size;
    const float* img   = (const float*)d_in[0];
    const float* img2  = (const float*)d_in[1];
    const float* cw1   = (const float*)d_in[2];
    const float* cw234 = (const float*)d_in[3];
    const float* cb    = (const float*)d_in[4];
    const float* bng   = (const float*)d_in[5];
    const float* bnb   = (const float*)d_in[6];
    const float* gw1   = (const float*)d_in[7];
    const float* gw234 = (const float*)d_in[8];
    const float* gb    = (const float*)d_in[9];
    const float* fw1   = (const float*)d_in[10];
    const float* fb1   = (const float*)d_in[11];
    const float* fc2w  = (const float*)d_in[12];
    const float* fc2b  = (const float*)d_in[13];
    const float* fc3w  = (const float*)d_in[14];
    const float* fc3b  = (const float*)d_in[15];

    char* ws = (char*)d_ws;
    size_t off = 0;
    auto alloc = [&](size_t bytes) -> char* {
        char* p = ws + off;
        off += (bytes + 255) & ~(size_t)255;
        return p;
    };

    float*    c1    = (float*)alloc(2457600ull * 4);   // 64x24x40x40
    float*    c2    = (float*)alloc(614400ull * 4);    // 64x24x20x20
    float*    c3    = (float*)alloc(153600ull * 4);    // 64x24x10x10
    float*    c4    = (float*)alloc(38400ull * 4);     // 64x24x5x5
    float*    stm   = (float*)alloc(64 * 4);
    float*    str   = (float*)alloc(64 * 4);
    float*    feat1 = (float*)alloc(41600ull * 4);     // [64,25,26]
    float*    feat2 = (float*)alloc(41600ull * 4);     // qst = feat2 flat [64,650]
    float*    Uc    = (float*)alloc(409600ull * 4);
    float*    Ua    = (float*)alloc(409600ull * 4);
    float*    Uq    = (float*)alloc(16384ull * 4);
    _Float16* gWh   = (_Float16*)alloc(196608ull * 2); // g_w234 in f16
    _Float16* hA    = (_Float16*)alloc(10240000ull * 2);
    _Float16* hB    = (_Float16*)alloc(10240000ull * 2);
    float*    h4f   = (float*)alloc(10240000ull * 4);
    float*    xg    = (float*)alloc(16384ull * 4);
    float*    xf    = (float*)alloc(16384ull * 4);
    float*    yb2   = (float*)alloc(16384ull * 4);
    float*    yb3   = (float*)alloc(8192ull * 4);

    auto g1d = [](int n) { return (n + 255) / 256; };

    auto run_cnn = [&](const float* im, float* featOut) {
        conv_relu_k<<<g1d(64*24*1600), 256, 0, stream>>>(im, cw1, cb + 0, c1, 3, 80, 80, 24, 40, 40);
        bn_stats_k<<<24, 256, 0, stream>>>(c1, stm, str, 24, 1600);
        bn_apply_k<<<g1d(64*24*1600), 256, 0, stream>>>(c1, stm, str, bng + 0, bnb + 0, 24, 1600);

        conv_relu_k<<<g1d(64*24*400), 256, 0, stream>>>(c1, cw234 + 0*5184, cb + 24, c2, 24, 40, 40, 24, 20, 20);
        bn_stats_k<<<24, 256, 0, stream>>>(c2, stm, str, 24, 400);
        bn_apply_k<<<g1d(64*24*400), 256, 0, stream>>>(c2, stm, str, bng + 24, bnb + 24, 24, 400);

        conv_relu_k<<<g1d(64*24*100), 256, 0, stream>>>(c2, cw234 + 1*5184, cb + 48, c3, 24, 20, 20, 24, 10, 10);
        bn_stats_k<<<24, 256, 0, stream>>>(c3, stm, str, 24, 100);
        bn_apply_k<<<g1d(64*24*100), 256, 0, stream>>>(c3, stm, str, bng + 48, bnb + 48, 24, 100);

        conv_relu_k<<<g1d(64*24*25), 256, 0, stream>>>(c3, cw234 + 2*5184, cb + 72, c4, 24, 10, 10, 24, 5, 5);
        bn_stats_k<<<24, 256, 0, stream>>>(c4, stm, str, 24, 25);
        bn_apply_k<<<g1d(64*24*25), 256, 0, stream>>>(c4, stm, str, bng + 72, bnb + 72, 24, 25);

        build_feat_k<<<g1d(64*25*26), 256, 0, stream>>>(c4, featOut);
    };

    run_cnn(img, feat1);
    run_cnn(img2, feat2);

    // g_fc1 via concat-split low-rank decomposition
    u_proj_k<<<g1d(409600), 256, 0, stream>>>(feat1, gw1, 0,  Uc);
    u_proj_k<<<g1d(409600), 256, 0, stream>>>(feat1, gw1, 26, Ua);
    uq_k<<<g1d(16384), 256, 0, stream>>>(feat2, gw1, Uq);
    h1_k<<<g1d(NPAIRROWS * 256), 256, 0, stream>>>(Uc, Ua, Uq, gb + 0, hA);

    // g_fc2..4 weights to f16, then WMMA GEMMs (40000x256 @ 256x256 each)
    f32_to_f16_k<<<g1d(196608), 256, 0, stream>>>(gw234, gWh, 196608);

    dim3 gemmGrid((2500 + 7) / 8, 4);   // mTile groups of 8 x four 64-col strips
    gemm_wmma_k<<<gemmGrid, 256, 0, stream>>>(hA, gWh + 0,      gb + 256, hB, nullptr, NPAIRROWS);
    gemm_wmma_k<<<gemmGrid, 256, 0, stream>>>(hB, gWh + 65536,  gb + 512, hA, nullptr, NPAIRROWS);
    gemm_wmma_k<<<gemmGrid, 256, 0, stream>>>(hA, gWh + 131072, gb + 768, nullptr, h4f, NPAIRROWS);

    // deterministic sum-pool over 625 pairs
    pool_k<<<g1d(16384), 256, 0, stream>>>(h4f, xg);

    // f network + grouped softmax
    fc_k<<<g1d(16384), 256, 0, stream>>>(xg,  fw1,  fb1,  xf,  64, 256, 256, 1);
    fc_k<<<g1d(16384), 256, 0, stream>>>(xf,  fc2w, fc2b, yb2, 64, 256, 256, 1);
    fc_k<<<g1d(8192),  256, 0, stream>>>(yb2, fc3w, fc3b, yb3, 64, 256, 128, 0);
    softmax8_k<<<g1d(1024), 256, 0, stream>>>(yb3, (float*)d_out);
}
